// Observer_4226247819522
// MI455X (gfx1250) — matrix operand order repaired and checked by reading, not compiled, not run on hardware
//
#include <hip/hip_runtime.h>
#include <hip/hip_bf16.h>

// ---------------------------------------------------------------------------
// gfx1250 (CDNA5, wave32): bf16 WMMA GEMM with async global->LDS tile loads.
// Block tile 64(M)x128(N), 4 waves, wave tile 32x64 (8 WMMAs / K-step of 32).
// ---------------------------------------------------------------------------
typedef __bf16 v4bf  __attribute__((ext_vector_type(4)));
typedef __bf16 v8bf  __attribute__((ext_vector_type(8)));
typedef __bf16 v16bf __attribute__((ext_vector_type(16)));
typedef float  v8f   __attribute__((ext_vector_type(8)));

#define NEG_INF (-1e9f)

// CDNA5 async global->LDS copy (ASYNCcnt-tracked). 16 bytes per lane.
#define ASYNC_LOAD_B128(ldsoff, gaddr)                                         \
  asm volatile("global_load_async_to_lds_b128 %0, %1, off" ::"v"(ldsoff),      \
               "v"(gaddr)                                                      \
               : "memory")
#define WAIT_ASYNC_0 asm volatile("s_wait_asynccnt 0x0" ::: "memory")
#define WAIT_ASYNC_6 asm volatile("s_wait_asynccnt 0x6" ::: "memory")

__device__ __forceinline__ unsigned lds_addr(const void* p) {
  return (unsigned)(unsigned long long)p;  // low 32 bits = LDS offset
}

// A fragment, bf16 16x32 (ISA 7.12.2): lanes 0-15 -> row m=lane, K {0..7,16..23};
// lanes 16-31 -> row m=lane-16, K {8..15,24..31}. LDS tile [64 M][32 K].
__device__ __forceinline__ v16bf load_a_frag(const __bf16* lA, int mBase, int lane) {
  const int half = lane >> 4;
  const int m    = lane & 15;
  const __bf16* p = lA + (mBase + m) * 32 + half * 8;
  v8bf lo = *(const v8bf*)(p);
  v8bf hi = *(const v8bf*)(p + 16);
  v16bf f;
#pragma unroll
  for (int j = 0; j < 8; ++j) { f[j] = lo[j]; f[8 + j] = hi[j]; }
  return f;
}

// B fragment: wmma-B = (B_source[N][K])^T; lane n<16 -> col n, K {0..7,16..23},
// lane>=16 -> col lane-16, K {8..15,24..31}. LDS tile [128 N][32 K].
__device__ __forceinline__ v16bf load_b_frag(const __bf16* lB, int nBase, int lane) {
  const int half = lane >> 4;
  const int n    = lane & 15;
  const __bf16* p = lB + (nBase + n) * 32 + half * 8;
  v8bf lo = *(const v8bf*)(p);
  v8bf hi = *(const v8bf*)(p + 16);
  v16bf f;
#pragma unroll
  for (int j = 0; j < 8; ++j) { f[j] = lo[j]; f[8 + j] = hi[j]; }
  return f;
}

// ---------------------------------------------------------------------------
// C[M,N] = alpha * A[M,K] @ B[N,K]^T (+ optional causal mask).  B always bf16.
// A_BF16: A bf16 (async, double-buffered) else fp32 (convert on load).
// GATHER_B: B row r -> gather[r/room]*room + r%room.
// CAUSAL:  epilogue mask; blocks fully above the diagonal skip the K-loop.
// TRIL:    A[m,k] == 0 for k > m  -> trim K-loop to (m0+64)/32 tiles.
// OUT: 0 = fp32 C[m,n]; 1 = bf16 C[m,n]; 2 = bf16 transposed C[n,m].
// grid.z: A += z*strideAz, B += (z%zmodB)*strideBz, C += z*strideCz.
// ---------------------------------------------------------------------------
template <bool A_BF16, bool GATHER_B, bool CAUSAL, bool TRIL, int OUT>
__global__ __launch_bounds__(128)
void gemm_wmma(const void* __restrict__ Av, int lda, long long strideAz,
               const __bf16* __restrict__ B, int ldb, long long strideBz, int zmodB,
               void* __restrict__ Cv, int ldc, long long strideCz,
               int K, float alpha,
               const long long* __restrict__ gather, int room) {
  __shared__ __bf16 lA[2][64 * 32];    //  8 KB
  __shared__ __bf16 lB[2][128 * 32];   // 16 KB

  const int tid  = threadIdx.x;
  const int wave = tid >> 5;
  const int lane = tid & 31;
  const int wm   = (wave >> 1) * 32;   // wave row offset in 64x128 block tile
  const int wn   = (wave & 1) * 64;    // wave col offset
  const int m0   = blockIdx.y * 64;
  const int n0   = blockIdx.x * 128;
  const int nk   = K / 32;
  // TRIL: weights are exactly zero for k > m -> only (m0+64)/32 K-tiles matter.
  const int nkEff = TRIL ? ((m0 + 64) / 32 < nk ? (m0 + 64) / 32 : nk) : nk;
  // CAUSAL: block entirely above the diagonal -> every element masked.
  const bool fullyMasked = CAUSAL && (n0 > m0 + 63);

  // A tile fill: row lr (0..63), 16-element chunk lh.  B tile fill: row tid.
  const int lr = tid >> 1;
  const int lh = tid & 1;

  long long brow = n0 + tid;
  if (GATHER_B) brow = gather[brow / room] * room + brow % room;

  const __bf16* aP  = (const __bf16*)Av + (long long)blockIdx.z * strideAz +
                      (long long)(m0 + lr) * lda + lh * 16;
  const float*  aPf = (const float*)Av + (long long)blockIdx.z * strideAz +
                      (long long)(m0 + lr) * lda + lh * 16;
  const __bf16* bP  = B + (long long)(blockIdx.z % zmodB) * strideBz +
                      brow * (long long)ldb;

  v8f acc[2][4];
#pragma unroll
  for (int tm = 0; tm < 2; ++tm)
#pragma unroll
    for (int tn = 0; tn < 4; ++tn)
#pragma unroll
      for (int i = 0; i < 8; ++i) acc[tm][tn][i] = 0.0f;

  auto issueA = [&](int t, int b) {   // 2 x b128 per thread (64x32 tile)
    unsigned o = lds_addr(&lA[b][lr * 32 + lh * 16]);
    unsigned long long g = (unsigned long long)(aP + (long long)t * 32);
    ASYNC_LOAD_B128(o, g);
    ASYNC_LOAD_B128(o + 16u, g + 16ull);
  };
  auto issueB = [&](int t, int b) {   // 4 x b128 per thread (128x32 tile)
    unsigned o = lds_addr(&lB[b][tid * 32]);
    unsigned long long g = (unsigned long long)(bP + (long long)t * 32);
    ASYNC_LOAD_B128(o, g);
    ASYNC_LOAD_B128(o + 16u, g + 16ull);
    ASYNC_LOAD_B128(o + 32u, g + 32ull);
    ASYNC_LOAD_B128(o + 48u, g + 48ull);
  };
  auto compute = [&](int b) {
    v16bf af[2], bf[4];
#pragma unroll
    for (int tm = 0; tm < 2; ++tm) af[tm] = load_a_frag(lA[b], wm + tm * 16, lane);
#pragma unroll
    for (int tn = 0; tn < 4; ++tn) bf[tn] = load_b_frag(lB[b], wn + tn * 16, lane);
#pragma unroll
    for (int tm = 0; tm < 2; ++tm)
#pragma unroll
      for (int tn = 0; tn < 4; ++tn)
        acc[tm][tn] = __builtin_amdgcn_wmma_f32_16x16x32_bf16(
            false, af[tm], false, bf[tn], (short)0, acc[tm][tn], false, false);
  };

  if (!fullyMasked) {
    if (A_BF16) {
      // Fully async, double-buffered: fill of tile t+1 overlaps WMMAs of tile t.
      issueA(0, 0);
      issueB(0, 0);
      for (int t = 0; t < nkEff; ++t) {
        const int b = t & 1;
        if (t + 1 < nkEff) {
          issueA(t + 1, b ^ 1);
          issueB(t + 1, b ^ 1);
          WAIT_ASYNC_6;        // 6 oldest (tile t) complete; tile t+1 in flight
        } else {
          WAIT_ASYNC_0;
        }
        __syncthreads();        // tile t visible to all waves
        compute(b);
        __syncthreads();        // all waves done reading before buffer reuse
      }
    } else {
      // fp32 A (attention weights): convert on load; B still async.
      for (int t = 0; t < nkEff; ++t) {
        __syncthreads();
        issueB(t, 0);
        {
          const float* s = aPf + (long long)t * 32;
          float4 f0 = *(const float4*)(s);
          float4 f1 = *(const float4*)(s + 4);
          float4 f2 = *(const float4*)(s + 8);
          float4 f3 = *(const float4*)(s + 12);
          v8bf p0, p1;
          p0[0] = (__bf16)f0.x; p0[1] = (__bf16)f0.y; p0[2] = (__bf16)f0.z; p0[3] = (__bf16)f0.w;
          p0[4] = (__bf16)f1.x; p0[5] = (__bf16)f1.y; p0[6] = (__bf16)f1.z; p0[7] = (__bf16)f1.w;
          p1[0] = (__bf16)f2.x; p1[1] = (__bf16)f2.y; p1[2] = (__bf16)f2.z; p1[3] = (__bf16)f2.w;
          p1[4] = (__bf16)f3.x; p1[5] = (__bf16)f3.y; p1[6] = (__bf16)f3.z; p1[7] = (__bf16)f3.w;
          *(v8bf*)(&lA[0][lr * 32 + lh * 16])     = p0;
          *(v8bf*)(&lA[0][lr * 32 + lh * 16 + 8]) = p1;
        }
        WAIT_ASYNC_0;
        __syncthreads();
        compute(0);
      }
    }
  }

  // Epilogue. C layout: VGPR i -> M = i + (lane>=16)*8, N = lane&15.
  // For fullyMasked blocks acc==0 and every (row,col) is masked -> writes -1e9.
  float*  Cf = (float*)Cv + (long long)blockIdx.z * strideCz;
  __bf16* Cb = (__bf16*)Cv + (long long)blockIdx.z * strideCz;
#pragma unroll
  for (int tm = 0; tm < 2; ++tm)
#pragma unroll
    for (int tn = 0; tn < 4; ++tn) {
      const int rbase = m0 + wm + tm * 16 + (lane >> 4) * 8;
      const int col   = n0 + wn + tn * 16 + (lane & 15);
#pragma unroll
      for (int i = 0; i < 8; ++i) {
        const int row = rbase + i;
        float v = acc[tm][tn][i] * alpha;
        if (CAUSAL && col > row) v += NEG_INF;
        if (OUT == 0)      Cf[(long long)row * ldc + col] = v;
        else if (OUT == 1) Cb[(long long)row * ldc + col] = (__bf16)v;
        else               Cb[(long long)col * ldc + row] = (__bf16)v;  // transposed
      }
    }
}

// ---------------------------------------------------------------------------
// fp32 -> bf16 elementwise convert (vectorized, n multiple of 4).
// ---------------------------------------------------------------------------
__global__ __launch_bounds__(256)
void cvt_f32_bf16(const float* __restrict__ s, __bf16* __restrict__ d, long long n4) {
  long long i = (long long)blockIdx.x * 256 + threadIdx.x;
  if (i < n4) {
    float4 f = ((const float4*)s)[i];
    v4bf o;
    o[0] = (__bf16)f.x; o[1] = (__bf16)f.y; o[2] = (__bf16)f.z; o[3] = (__bf16)f.w;
    ((v4bf*)d)[i] = o;
  }
}

// ---------------------------------------------------------------------------
// OT[n][t] = (bf16) src[gather(t)][n]  — tiled transpose + convert + row gather.
// ---------------------------------------------------------------------------
__global__ __launch_bounds__(256)
void transpose_gather_cvt(const float* __restrict__ src, __bf16* __restrict__ dst,
                          const long long* __restrict__ gather, int room, int n) {
  __shared__ float tile[32][33];
  const int tBase = blockIdx.y * 32;
  const int nBase = blockIdx.x * 32;
#pragma unroll
  for (int j = threadIdx.y; j < 32; j += 8) {
    long long t = tBase + j;
    long long g = gather[t / room] * room + t % room;
    tile[j][threadIdx.x] = src[g * n + nBase + threadIdx.x];
  }
  __syncthreads();
#pragma unroll
  for (int j = threadIdx.y; j < 32; j += 8) {
    dst[(long long)(nBase + j) * n + tBase + threadIdx.x] = (__bf16)tile[threadIdx.x][j];
  }
}

// ---------------------------------------------------------------------------
// In-place causal row softmax: one 256-thread block per row.  Columns beyond
// the diagonal are known-masked: reduce only over c <= row, write zeros after.
// ---------------------------------------------------------------------------
__global__ __launch_bounds__(256)
void softmax_rows_causal(float* __restrict__ data, int ncols) {
  __shared__ float red[256];
  const int tid = threadIdx.x;
  const int row = blockIdx.x % ncols;   // row index within this head
  const int lim = row + 1;              // valid prefix length
  float* p = data + (long long)blockIdx.x * ncols;

  float m = -3.4e38f;
  for (int c = tid; c < lim; c += 256) m = fmaxf(m, p[c]);
  red[tid] = m; __syncthreads();
  for (int s = 128; s > 0; s >>= 1) {
    if (tid < s) red[tid] = fmaxf(red[tid], red[tid + s]);
    __syncthreads();
  }
  const float rowmax = red[0]; __syncthreads();

  float sum = 0.0f;
  for (int c = tid; c < lim; c += 256) {
    float e = __expf(p[c] - rowmax);
    p[c] = e;
    sum += e;
  }
  red[tid] = sum; __syncthreads();
  for (int s = 128; s > 0; s >>= 1) {
    if (tid < s) red[tid] += red[tid + s];
    __syncthreads();
  }
  const float inv = 1.0f / red[0];
  for (int c = tid; c < lim; c += 256) p[c] *= inv;
  for (int c = lim + tid; c < ncols; c += 256) p[c] = 0.0f;
}

// ---------------------------------------------------------------------------
// Orchestration
// ---------------------------------------------------------------------------
extern "C" void kernel_launch(void* const* d_in, const int* in_sizes, int n_in,
                              void* d_out, int out_size, void* d_ws, size_t ws_size,
                              hipStream_t stream) {
  const float*     hs    = (const float*)d_in[0];      // [2048, 2048]
  const float*     qw    = (const float*)d_in[1];      // [2048, 2048]
  const float*     kw    = (const float*)d_in[2];      // [256, 2048]
  const float*     vw    = (const float*)d_in[3];      // [256, 2048]
  const float*     ow    = (const float*)d_in[4];      // [2048, 2048]
  const long long* rooms = (const long long*)d_in[5];  // [8] int64

  const int S = 2048, H = 2048, ROOM = 256, NH = 16, HD = 128;
  const long long SH = (long long)S * H;      // 4M
  const long long SR = (long long)S * ROOM;   // 0.5M
  const long long RH = (long long)ROOM * H;   // 0.5M
  const float inv_sqrt_hd = 0.08838834764831845f;  // 1/sqrt(128)

  // bf16 workspace (44 MB total)
  __bf16* hsb = (__bf16*)d_ws;       // [S, H]
  __bf16* qwb = hsb + SH;            // [H, H]
  __bf16* kwb = qwb + SH;            // [ROOM, H]
  __bf16* vwb = kwb + RH;            // [ROOM, H]
  __bf16* OT  = vwb + RH;            // [H, H]   = o_active^T
  __bf16* Qb  = OT + SH;             // [S, H]
  __bf16* Kb  = Qb + SH;             // [S, ROOM]
  __bf16* Vt  = Kb + SR;             // [ROOM, S] = Vp^T
  __bf16* AOb = Vt + SR;             // [S, H]
  float*  Oout = (float*)d_out;      // [S, H]
  float*  Wt   = Oout + SH;          // [NH, S, S] attn weights (fp32 output)

  dim3 blk(128);

  // 0) one-time bf16 conversions of all GEMM operands
  cvt_f32_bf16<<<(int)(SH / 4 / 256), 256, 0, stream>>>(hs, hsb, SH / 4);
  cvt_f32_bf16<<<(int)(SH / 4 / 256), 256, 0, stream>>>(qw, qwb, SH / 4);
  cvt_f32_bf16<<<(int)(RH / 4 / 256), 256, 0, stream>>>(kw, kwb, RH / 4);
  cvt_f32_bf16<<<(int)(RH / 4 / 256), 256, 0, stream>>>(vw, vwb, RH / 4);
  transpose_gather_cvt<<<dim3(H / 32, H / 32), dim3(32, 8), 0, stream>>>(
      ow, OT, rooms, ROOM, H);

  // 1) Q = hs @ q_block^T (gathered rows)  -> bf16
  gemm_wmma<true, true, false, false, 1><<<dim3(H / 128, S / 64, 1), blk, 0, stream>>>(
      hsb, H, 0, qwb, H, 0, 1, Qb, H, 0, H, 1.0f, rooms, ROOM);

  // 2) Kp = hs @ k_block^T -> bf16 ; Vt = (hs @ v_block^T)^T -> bf16 [ROOM, S]
  gemm_wmma<true, false, false, false, 1><<<dim3(ROOM / 128, S / 64, 1), blk, 0, stream>>>(
      hsb, H, 0, kwb, H, 0, 1, Kb, ROOM, 0, H, 1.0f, nullptr, ROOM);
  gemm_wmma<true, false, false, false, 2><<<dim3(ROOM / 128, S / 64, 1), blk, 0, stream>>>(
      hsb, H, 0, vwb, H, 0, 1, Vt, S, 0, H, 1.0f, nullptr, ROOM);

  // 3) scores_h = (Qh @ Kh^T)/sqrt(hd) + causal -> fp32 into d_out; blocks
  //    fully above the diagonal skip their K-loop entirely.
  gemm_wmma<true, false, true, false, 0><<<dim3(S / 128, S / 64, NH), blk, 0, stream>>>(
      Qb, H, HD, Kb, ROOM, HD, 2, Wt, S, (long long)S * S, HD, inv_sqrt_hd,
      nullptr, ROOM);

  // 4) causal softmax rows in place (NH*S rows)
  softmax_rows_causal<<<NH * S, 256, 0, stream>>>(Wt, S);

  // 5) attn_out_h = W_h @ Vh  (TRIL: weights zero for k > m -> K-loop trimmed)
  gemm_wmma<false, false, false, true, 1><<<dim3(HD / 128, S / 64, NH), blk, 0, stream>>>(
      Wt, S, (long long)S * S, Vt, S, (long long)HD * S, 2, AOb, H, HD, S, 1.0f,
      nullptr, ROOM);

  // 6) o_out = attn_out @ o_active = AOb @ OT^T  -> fp32 d_out
  gemm_wmma<true, false, false, false, 0><<<dim3(H / 128, S / 64, 1), blk, 0, stream>>>(
      AOb, H, 0, OT, H, 0, 1, Oout, H, 0, H, 1.0f, nullptr, ROOM);
}
